// CrossAttention_14207751815513
// MI455X (gfx1250) — matrix operand order repaired and checked by reading, not compiled, not run on hardware
//
#include <hip/hip_runtime.h>
#include <hip/hip_bf16.h>
#include <stdint.h>
#include <stddef.h>

// ---------------------------------------------------------------------------
// CrossAttention, single query per head, algebraically collapsed:
//   logits[h,s] = (Wk_h^T q_h) . x2[s]      (k never materialized)
//   out[h,:]    = Wv_h (sum_s p[h,s] x2[s]) (v never materialized)
// Two thin bf16 WMMA GEMM passes over x2 dominate; x2 (134MB) < L2 (192MB),
// so the second pass is L2-resident.  Small operands (wkt / p-chunk) are
// staged in LDS so the VMEM pipe carries only the compulsory x2 stream.
// ---------------------------------------------------------------------------

typedef __bf16 v16bf __attribute__((ext_vector_type(16)));
typedef __bf16 v8bf  __attribute__((ext_vector_type(8)));
typedef __bf16 v4bf  __attribute__((ext_vector_type(4)));
typedef float  v8f   __attribute__((ext_vector_type(8)));
typedef float  v4f   __attribute__((ext_vector_type(4)));

#define IN_DIM   2048
#define SEQ      16384
#define HEADS    16
#define KD       128
#define OUT_DIM  512
#define SPLITK   16
#define SCHUNK   (SEQ / SPLITK)         // 1024 rows per split-K chunk
#define SCALE    0.08838834764831845f   // 1/sqrt(128)

// f32 -> bf16: let clang pick the hardware conversion (v_cvt_*bf16_f32)
static __device__ __forceinline__ __bf16 f2bf(float f) { return (__bf16)f; }

static __device__ __forceinline__ float wave_sum(float v) {
    #pragma unroll
    for (int off = 16; off; off >>= 1) v += __shfl_xor(v, off, 32);
    return v;
}

// --- K1: q[i] = (Wq[i,:] . x1) * 1/sqrt(K).  One wave32 per output row. ----
__global__ void __launch_bounds__(256) k_qproj(const float* __restrict__ Wq,
                                               const float* __restrict__ x1,
                                               float* __restrict__ q) {
    int wave = threadIdx.x >> 5, lane = threadIdx.x & 31;
    int row  = blockIdx.x * 8 + wave;                 // 2048 rows
    const float* wrow = Wq + (size_t)row * IN_DIM;
    float acc = 0.f;
    for (int c = lane; c < IN_DIM; c += 32) acc += wrow[c] * x1[c];
    acc = wave_sum(acc);
    if (lane == 0) q[row] = acc * SCALE;
}

// --- K2: wkt[c][h] = sum_j q[h*128+j] * Wk[h*128+j][c]   (bf16, transposed
//     so each WMMA B row = 16 contiguous bf16 = one 32B load) --------------
__global__ void __launch_bounds__(256) k_fold_wk(const float* __restrict__ Wk,
                                                 const float* __restrict__ q,
                                                 __bf16* __restrict__ wkt) {
    int t = blockIdx.x * blockDim.x + threadIdx.x;    // 32768 = 16*2048
    int h = t >> 11, c = t & 2047;
    const float* base = Wk + (size_t)h * KD * IN_DIM + c;
    const float* qh   = q + h * KD;
    float acc = 0.f;
    #pragma unroll 4
    for (int j = 0; j < KD; ++j) acc += qh[j] * base[(size_t)j * IN_DIM];
    wkt[(size_t)c * HEADS + h] = f2bf(acc);
}

// --- K3: logits = x2 * wkt  via v_wmma_f32_16x16x32_bf16.
//     One wave per 16-row x2 tile, 64 K-steps over the 2048 inner dim.
//     wkt (64KB) is staged once per block in LDS; the global loop reads
//     only the compulsory x2 stream. ---------------------------------------
__global__ void __launch_bounds__(256) k_logits(const float* __restrict__ x2,
                                                const __bf16* __restrict__ wkt,
                                                float* __restrict__ logits) {
    __shared__ __align__(32) __bf16 swkt[IN_DIM * HEADS];   // 64 KB
    {   // cooperative 64KB copy: 4096 uint4, 16 per thread
        const uint4* src = (const uint4*)wkt;
        uint4* dst = (uint4*)swkt;
        #pragma unroll
        for (int i = 0; i < 16; ++i) dst[threadIdx.x + 256 * i] = src[threadIdx.x + 256 * i];
    }
    __syncthreads();

    int wave = threadIdx.x >> 5, lane = threadIdx.x & 31;
    int m = lane & 15, hi = lane >> 4;
    int s0 = (blockIdx.x * 8 + wave) * 16;            // 128 blocks * 8 waves * 16 = 16384
    const float* arow = x2 + (size_t)(s0 + m) * IN_DIM;
    const v16bf* bp = (const v16bf*)swkt;             // row k = 16 bf16 (heads)
    v8f acc = {};
    #pragma unroll 2
    for (int kk = 0; kk < IN_DIM; kk += 32) {
        __builtin_prefetch(arow + kk + 256, 0, 0);    // global_prefetch on x2 stream
        // A fragment (16x32 bf16): lane m holds K = {hi*8..hi*8+7, 16+hi*8..}
        const v4f* a4 = (const v4f*)(arow + kk + hi * 8);
        v4bf a0 = __builtin_convertvector(a4[0], v4bf);
        v4bf a1 = __builtin_convertvector(a4[1], v4bf);
        const v4f* a5 = (const v4f*)(arow + kk + 16 + hi * 8);
        v4bf a2 = __builtin_convertvector(a5[0], v4bf);
        v4bf a3 = __builtin_convertvector(a5[1], v4bf);
        v16bf A;
        #pragma unroll
        for (int e = 0; e < 4; ++e) {
            A[e] = a0[e];  A[4 + e] = a1[e];  A[8 + e] = a2[e];  A[12 + e] = a3[e];
        }
        // B fragment (32x16 bf16) from LDS: lane&15 = N (head), elements = K (table order)
        v16bf B;
        #pragma unroll
        for (int e = 0; e < 8; ++e) {
            B[e] = swkt[(kk + hi * 8 + e) * HEADS + m];  B[8 + e] = swkt[(kk + 16 + hi * 8 + e) * HEADS + m];
        }
        acc = __builtin_amdgcn_wmma_f32_16x16x32_bf16(false, A, false, B,
                                                      (short)0, acc, false, false);
    }
    // D layout: VGPR r, M = r + 8*hi, N = lane&15.  Store logits[head][s].
    #pragma unroll
    for (int r = 0; r < 8; ++r) {
        int M = r + hi * 8, N = lane & 15;
        logits[(size_t)N * SEQ + s0 + M] = acc[r];
    }
}

// --- K4: per-head softmax over 16384 logits -> bf16 probabilities ----------
__global__ void __launch_bounds__(256) k_softmax(const float* __restrict__ logits,
                                                 __bf16* __restrict__ p) {
    __shared__ float red[256];
    int h = blockIdx.x;
    const float* row = logits + (size_t)h * SEQ;
    float m = -3.4e38f;
    for (int s = threadIdx.x; s < SEQ; s += 256) m = fmaxf(m, row[s]);
    red[threadIdx.x] = m; __syncthreads();
    for (int off = 128; off; off >>= 1) {
        if (threadIdx.x < off) red[threadIdx.x] = fmaxf(red[threadIdx.x], red[threadIdx.x + off]);
        __syncthreads();
    }
    m = red[0]; __syncthreads();
    float sum = 0.f;
    for (int s = threadIdx.x; s < SEQ; s += 256) sum += __expf(row[s] - m);
    red[threadIdx.x] = sum; __syncthreads();
    for (int off = 128; off; off >>= 1) {
        if (threadIdx.x < off) red[threadIdx.x] += red[threadIdx.x + off];
        __syncthreads();
    }
    float inv = 1.f / red[0];
    __bf16* prow = p + (size_t)h * SEQ;
    for (int s = threadIdx.x; s < SEQ; s += 256) prow[s] = f2bf(__expf(row[s] - m) * inv);
}

// --- K5: part[bk] = p[16 x Schunk] * x2[Schunk x 16cols]  (split-K WMMA).
//     The 32KB p-chunk is staged once per block in LDS (shared by 8 waves).
__global__ void __launch_bounds__(256) k_av(const float* __restrict__ x2,
                                            const __bf16* __restrict__ p,
                                            float* __restrict__ part) {
    __shared__ __align__(32) __bf16 sp[HEADS * SCHUNK];     // 32 KB
    int bk = blockIdx.y;                              // SPLITK chunks
    int sbase = bk * SCHUNK;                          // 1024 rows per chunk
    {   // cooperative 32KB copy: 2048 uint4, 8 per thread; row = 128 uint4
        uint4* dst = (uint4*)sp;
        #pragma unroll
        for (int i = 0; i < 8; ++i) {
            int idx = threadIdx.x + 256 * i;          // 0..2047
            int r = idx >> 7, c = idx & 127;
            dst[idx] = ((const uint4*)(p + (size_t)r * SEQ + sbase))[c];
        }
    }
    __syncthreads();

    int wave = threadIdx.x >> 5, lane = threadIdx.x & 31;
    int m = lane & 15, hi = lane >> 4;
    int c0 = (blockIdx.x * 8 + wave) * 16;            // 16 blocks * 8 waves * 16 = 2048 cols
    const __bf16* prow = sp + (size_t)m * SCHUNK;     // A row = head m (LDS)
    v8f acc = {};
    #pragma unroll 2
    for (int kk = 0; kk < SCHUNK; kk += 32) {         // 32 WMMA steps
        // A fragment from LDS: probabilities (bf16), standard K striping
        v8bf c1 = *(const v8bf*)(prow + kk + hi * 8);
        v8bf c2 = *(const v8bf*)(prow + kk + 16 + hi * 8);
        v16bf A;
        #pragma unroll
        for (int e = 0; e < 8; ++e) { A[e] = c1[e]; A[8 + e] = c2[e]; }
        // B fragment: lane&15 = N (col c0+m), elements = K rows (table order)
        const float* xrow = x2 + (size_t)(sbase + kk + hi * 8) * IN_DIM + c0 + m;
        __builtin_prefetch(xrow + 32 * IN_DIM, 0, 0);
        v16bf B;
        #pragma unroll
        for (int e = 0; e < 8; ++e) {
            B[e] = f2bf(xrow[(size_t)e * IN_DIM]);  B[8 + e] = f2bf(xrow[(size_t)(16 + e) * IN_DIM]);
        }
        acc = __builtin_amdgcn_wmma_f32_16x16x32_bf16(false, A, false, B,
                                                      (short)0, acc, false, false);
    }
    #pragma unroll
    for (int r = 0; r < 8; ++r) {
        int M = r + hi * 8, N = lane & 15;            // M = head, N = col offset
        part[((size_t)bk * HEADS + M) * IN_DIM + c0 + N] = acc[r];
    }
}

// --- K6: y = sum over split-K partials (deterministic, no float atomics) ---
__global__ void __launch_bounds__(256) k_reduce_y(const float* __restrict__ part,
                                                  float* __restrict__ y) {
    int t = blockIdx.x * blockDim.x + threadIdx.x;    // 32768 = 16*2048
    float acc = 0.f;
    #pragma unroll
    for (int bk = 0; bk < SPLITK; ++bk) acc += part[(size_t)bk * HEADS * IN_DIM + t];
    y[t] = acc;
}

// --- K7: out_attn[h*128+v] = Wv[h*128+v,:] . y[h,:]  (wave-per-row GEMV) ---
__global__ void __launch_bounds__(256) k_vproj(const float* __restrict__ Wv,
                                               const float* __restrict__ y,
                                               float* __restrict__ oat) {
    int wave = threadIdx.x >> 5, lane = threadIdx.x & 31;
    int row = blockIdx.x * 8 + wave;                  // 2048 rows
    int h = row >> 7;
    const float* wrow = Wv + (size_t)row * IN_DIM;
    const float* yrow = y + (size_t)h * IN_DIM;
    float acc = 0.f;
    for (int c = lane; c < IN_DIM; c += 32) acc += wrow[c] * yrow[c];
    acc = wave_sum(acc);
    if (lane == 0) oat[row] = acc;
}

// --- K8: final = Wo * out_attn + bo ----------------------------------------
__global__ void __launch_bounds__(256) k_out(const float* __restrict__ Wo,
                                             const float* __restrict__ bo,
                                             const float* __restrict__ oat,
                                             float* __restrict__ out) {
    int wave = threadIdx.x >> 5, lane = threadIdx.x & 31;
    int row = blockIdx.x * 8 + wave;                  // 512 rows
    const float* wrow = Wo + (size_t)row * (HEADS * KD);
    float acc = 0.f;
    for (int c = lane; c < HEADS * KD; c += 32) acc += wrow[c] * oat[c];
    acc = wave_sum(acc);
    if (lane == 0) out[row] = acc + bo[row];
}

// ---------------------------------------------------------------------------
// Workspace layout (bytes)
static constexpr size_t OFF_Q      = 0;                                     // 2048 f32
static constexpr size_t OFF_WKT    = 8 * 1024;                              // 2048*16 bf16
static constexpr size_t OFF_LOGITS = OFF_WKT + (size_t)IN_DIM * HEADS * 2;  // 16*16384 f32
static constexpr size_t OFF_P      = OFF_LOGITS + (size_t)HEADS * SEQ * 4;  // 16*16384 bf16
static constexpr size_t OFF_PART   = OFF_P + (size_t)HEADS * SEQ * 2;       // 16*16*2048 f32
static constexpr size_t OFF_Y      = OFF_PART + (size_t)SPLITK * HEADS * IN_DIM * 4;
static constexpr size_t OFF_OATT   = OFF_Y + (size_t)HEADS * IN_DIM * 4;    // 2048 f32

extern "C" void kernel_launch(void* const* d_in, const int* in_sizes, int n_in,
                              void* d_out, int out_size, void* d_ws, size_t ws_size,
                              hipStream_t stream) {
    (void)in_sizes; (void)n_in; (void)out_size; (void)ws_size;
    const float* x1 = (const float*)d_in[0];
    const float* x2 = (const float*)d_in[1];
    const float* Wq = (const float*)d_in[2];
    const float* Wk = (const float*)d_in[3];
    const float* Wv = (const float*)d_in[4];
    const float* Wo = (const float*)d_in[5];
    const float* bo = (const float*)d_in[6];

    char* ws = (char*)d_ws;
    float*  q      = (float*)(ws + OFF_Q);
    __bf16* wkt    = (__bf16*)(ws + OFF_WKT);
    float*  logits = (float*)(ws + OFF_LOGITS);
    __bf16* p      = (__bf16*)(ws + OFF_P);
    float*  part   = (float*)(ws + OFF_PART);
    float*  y      = (float*)(ws + OFF_Y);
    float*  oat    = (float*)(ws + OFF_OATT);
    float*  out    = (float*)d_out;

    k_qproj   <<<256, 256, 0, stream>>>(Wq, x1, q);
    k_fold_wk <<<128, 256, 0, stream>>>(Wk, q, wkt);
    k_logits  <<<128, 256, 0, stream>>>(x2, wkt, logits);
    k_softmax <<<HEADS, 256, 0, stream>>>(logits, p);
    k_av      <<<dim3(16, SPLITK), 256, 0, stream>>>(x2, p, part);
    k_reduce_y<<<128, 256, 0, stream>>>(part, y);
    k_vproj   <<<256, 256, 0, stream>>>(Wv, y, oat);
    k_out     <<<64, 256, 0, stream>>>(Wo, bo, oat, out);
}
